// Decoder_78649441124508
// MI455X (gfx1250) — matrix operand order repaired and checked by reading, not compiled, not run on hardware
//
#include <hip/hip_runtime.h>
#include <hip/hip_bf16.h>

// ---------------------------------------------------------------------------
// Tacotron2 decoder for MI455X (gfx1250, wave32, WMMA bf16 16x16x32).
// - All weights cast to bf16 once (L2-resident; 192MB L2 holds the ~32MB of
//   recurrent weights re-read 800x, so per-step GEMMs never touch HBM).
// - Every matmul via v_wmma_f32_16x16x32_bf16, f32 accumulation.
// - Staging global->LDS uses GLOBAL_LOAD_ASYNC_TO_LDS_B128 (16B per lane,
//   ASYNCcnt) instead of scalar load/store; s_wait_asynccnt 0 before barriers.
// - LSTM cell, BN, activations fused into GEMM epilogues; rnn_lin query GEMM
//   fused into the LSTM1 kernel (shares the A operand). 4 kernels per step.
// ---------------------------------------------------------------------------

#define USE_ASYNC 1

typedef __attribute__((ext_vector_type(16))) __bf16 v16bf;
typedef __attribute__((ext_vector_type(8)))  float  v8f;

static __device__ inline unsigned short f2bf_bits(float f) {
  union { float f; unsigned int u; } v; v.f = f;
  unsigned int u = v.u;
  unsigned int r = (u + 0x7FFFu + ((u >> 16) & 1u)) >> 16;  // RNE
  return (unsigned short)r;
}

static __device__ inline float sigm(float x) { return 1.0f / (1.0f + expf(-x)); }

static __device__ inline v8f wmma_bf(v16bf a, v16bf b, v8f c) {
  return __builtin_amdgcn_wmma_f32_16x16x32_bf16(false, a, false, b, (short)0, c,
                                                 false, false);
}

// 16-byte global -> LDS copy. Async DMA path (gfx1250 GLOBAL_LOAD_ASYNC_TO_LDS,
// tracked by ASYNCcnt) or plain VGPR round-trip fallback.
static __device__ inline void cp16(unsigned short* dst_lds, const unsigned short* src) {
#if USE_ASYNC
  unsigned lds = (unsigned)(unsigned long long)dst_lds;  // low 32 bits = LDS addr
  unsigned long long ga = (unsigned long long)src;
  asm volatile("global_load_async_to_lds_b128 %0, %1, off"
               :: "v"(lds), "v"(ga) : "memory");
#else
  *(uint4*)dst_lds = *(const uint4*)src;
#endif
}
static __device__ inline void zero16(unsigned short* dst_lds) {
  uint4 z; z.x = 0u; z.y = 0u; z.z = 0u; z.w = 0u;
  *(uint4*)dst_lds = z;
}
static __device__ inline void async_fence() {
#if USE_ASYNC
  asm volatile("s_wait_asynccnt 0x0" ::: "memory");
#endif
}

// A fragment: 16x32 (MxK) bf16 tile in LDS row-major [m][k] (stride 32).
// lanes 0-15 hold K=0..15, lanes 16-31 hold K=16..31 (matches load_b_frag).
static __device__ inline v16bf load_a_frag(const unsigned short* lds_a) {
  int lane = threadIdx.x & 31;
  int m = lane & 15;
  int hi = lane >> 4;
  v16bf r;
#pragma unroll
  for (int h = 0; h < 16; ++h) {
    int k = hi * 16 + h;
    r[h] = __builtin_bit_cast(__bf16, lds_a[m * 32 + k]);
  }
  return r;
}

// B fragment: 32x16 (KxN) from LDS W-tile rows [n][k] (16 rows x 32 k).
// This code's order (not ISA table): lanes 0-15 hold K=0..15, lanes 16-31 hold K=16..31.
static __device__ inline v16bf load_b_frag(const unsigned short* lds_w) {
  int lane = threadIdx.x & 31;
  int n = lane & 15;
  int kb = (lane >> 4) * 16;
  v16bf r;
#pragma unroll
  for (int h = 0; h < 16; ++h)
    r[h] = __builtin_bit_cast(__bf16, lds_w[n * 32 + kb + h]);
  return r;
}

// ------------------------------ small utils --------------------------------

__global__ void k_f2b(const float* __restrict__ s, unsigned short* __restrict__ d, int n) {
  int i = blockIdx.x * blockDim.x + threadIdx.x;
  if (i < n) d[i] = f2bf_bits(s[i]);
}
__global__ void k_zero_f(float* p, int n) {
  int i = blockIdx.x * blockDim.x + threadIdx.x;
  if (i < n) p[i] = 0.0f;
}
__global__ void k_zero_h(unsigned short* p, int n) {
  int i = blockIdx.x * blockDim.x + threadIdx.x;
  if (i < n) p[i] = 0;
}
__global__ void k_concat2(const float* __restrict__ a, int la,
                          const float* __restrict__ b, int lb,
                          unsigned short* __restrict__ dst, int rows) {
  int K = la + lb;
  int i = blockIdx.x * blockDim.x + threadIdx.x;
  if (i >= rows * K) return;
  int r = i / K, k = i % K;
  float v = (k < la) ? a[r * la + k] : b[r * lb + (k - la)];
  dst[i] = f2bf_bits(v);
}
__global__ void k_bnfold(const float* g, const float* be, const float* mu,
                         const float* va, float* sc, float* sh, int n) {
  int i = blockIdx.x * blockDim.x + threadIdx.x;
  if (i >= n) return;
  float s = g[i] * rsqrtf(va[i] + 1e-5f);
  sc[i] = s;
  sh[i] = be[i] - mu[i] * s;
}
// conv weight repack: src [Cout][Cin][5] f32 -> dst [Cout][5][Cin] bf16
__global__ void k_convrepack(const float* __restrict__ src, unsigned short* __restrict__ dst,
                             int Cout, int Cin) {
  int i = blockIdx.x * blockDim.x + threadIdx.x;
  if (i >= Cout * Cin * 5) return;
  int n = i / (5 * Cin), rem = i % (5 * Cin), tap = rem / Cin, c = rem % Cin;
  dst[i] = f2bf_bits(src[((size_t)n * Cin + c) * 5 + tap]);
}
// mels [B,80,T] f32 -> bf16 rows r=t*B+b
__global__ void k_mels_to_rows(const float* __restrict__ m, unsigned short* __restrict__ X,
                               int B, int C, int T) {
  int i = blockIdx.x * blockDim.x + threadIdx.x;
  if (i >= B * C * T) return;
  int c = i % C, r = i / C, b = r % B, t = r / B;
  X[i] = f2bf_bits(m[(size_t)b * C * T + (size_t)c * T + t]);
}
// mel [B,80,T] f32 -> bf16 rows r=b*T+t
__global__ void k_mel_to_post(const float* __restrict__ m, unsigned short* __restrict__ X,
                              int B, int C, int T) {
  int i = blockIdx.x * blockDim.x + threadIdx.x;
  if (i >= B * C * T) return;
  int c = i % C, r = i / C, t = r % T, b = r / T;
  X[i] = f2bf_bits(m[(size_t)b * C * T + (size_t)c * T + t]);
}
__global__ void k_residual(const float* __restrict__ m0, const float* __restrict__ pt,
                           float* __restrict__ o, int B, int C, int T) {
  int i = blockIdx.x * blockDim.x + threadIdx.x;
  if (i >= B * C * T) return;
  int t = i % T, r = i / T, c = r % C, b = r / C;
  o[i] = m0[i] + pt[((size_t)b * T + t) * C + c];
}

// --------------------------- generic WMMA GEMM -----------------------------
// out = act(A[M,K](bf16) @ W[N,K](bf16)^T + bias). block=256 (8 waves),
// block tile = 32m x 64n, wave tile = 16x16, K chunked by 32 via LDS.
__global__ void k_gemm(const unsigned short* __restrict__ A,
                       const unsigned short* __restrict__ W,
                       const float* __restrict__ bias,
                       float* __restrict__ outF, unsigned short* __restrict__ outB,
                       int M, int N, int K, int act) {
  __shared__ __align__(16) unsigned short sA[32 * 32];
  __shared__ __align__(16) unsigned short sW[64 * 32];
  int tid = threadIdx.x, wave = tid >> 5, lane = tid & 31;
  int mt = wave & 1, nt = wave >> 1;
  int m0 = blockIdx.y * 32, n0 = blockIdx.x * 64;
  v8f acc = {0.f, 0.f, 0.f, 0.f, 0.f, 0.f, 0.f, 0.f};
  int KC = (K + 31) >> 5;
  for (int kc = 0; kc < KC; ++kc) {
    int kb = kc << 5;
    if (kb + 32 <= K) {  // vector fast path: 16B per lane
      {
        int r = tid >> 2, off = (tid & 3) * 8;  // W: 64 rows x 4 vec
        int n = n0 + r;
        if (n < N) cp16(sW + tid * 8, W + (size_t)n * K + kb + off);
        else zero16(sW + tid * 8);
      }
      if (tid < 128) {  // A: 32 rows x 4 vec
        int r = tid >> 2, off = (tid & 3) * 8;
        int gm = m0 + r;
        if (gm < M) cp16(sA + tid * 8, A + (size_t)gm * K + kb + off);
        else zero16(sA + tid * 8);
      }
    } else {  // scalar tail (K not multiple of 32)
      for (int i = tid; i < 1024; i += 256) {
        int r = i >> 5, k = kb + (i & 31);
        sA[i] = (k < K && (m0 + r) < M) ? A[(size_t)(m0 + r) * K + k] : 0;
      }
      for (int i = tid; i < 2048; i += 256) {
        int r = i >> 5, k = kb + (i & 31);
        int n = n0 + r;
        sW[i] = (k < K && n < N) ? W[(size_t)n * K + k] : 0;
      }
    }
    async_fence();
    __syncthreads();
    v16bf a = load_a_frag(sA + mt * 512);
    v16bf b = load_b_frag(sW + nt * 512);
    acc = wmma_bf(a, b, acc);
    __syncthreads();
  }
  int gn = n0 + nt * 16 + (lane & 15);
  int mb = m0 + mt * 16 + ((lane >> 4) << 3);
  if (gn < N) {
#pragma unroll
    for (int v = 0; v < 8; ++v) {
      int gm = mb + v;
      if (gm < M) {
        float x = acc[v] + (bias ? bias[gn] : 0.0f);
        if (act == 1) x = fmaxf(x, 0.0f);
        else if (act == 2) x = tanhf(x);
        if (outF) outF[(size_t)gm * N + gn] = x;
        if (outB) outB[(size_t)gm * N + gn] = f2bf_bits(x);
      }
    }
  }
}

// ------------------------ fused LSTM gate GEMM + cell ----------------------
// A rows (b=0..31) = concat(seg0,seg1,seg2) bf16; all segment lengths are
// multiples of 32, so each K-chunk lives in exactly one segment (uniform).
// Blocks 0..63: 16 cells each, i/f/g/o gate tiles + fused cell update.
// Block 64 (when Wq): q = A @ Wq^T (rnn_lin shares the A operand).
__global__ void k_lstm_step(const unsigned short* __restrict__ s0, int l0,
                            const unsigned short* __restrict__ s1, int l1,
                            const unsigned short* __restrict__ s2, int l2,
                            const unsigned short* __restrict__ Wc, int K,
                            const float* __restrict__ bias,
                            float* __restrict__ cs, float* __restrict__ hs,
                            unsigned short* __restrict__ hbf_out,
                            const unsigned short* __restrict__ Wq,
                            float* __restrict__ qout) {
  __shared__ __align__(16) unsigned short sA[32 * 32];
  __shared__ __align__(16) unsigned short sW[128 * 32];
  __shared__ float gbuf[32 * 64];
  int tid = threadIdx.x, wave = tid >> 5, lane = tid & 31;
  bool isq = (blockIdx.x == 64);
  int c0 = blockIdx.x * 16;
  int l01 = l0 + l1;
  v8f acc0 = {0.f, 0.f, 0.f, 0.f, 0.f, 0.f, 0.f, 0.f};
  v8f acc1 = {0.f, 0.f, 0.f, 0.f, 0.f, 0.f, 0.f, 0.f};
  int KC = (K + 31) >> 5;
  for (int kc = 0; kc < KC; ++kc) {
    int kb = kc << 5;
    // uniform segment select for this chunk
    const unsigned short* segp;
    int stride;
    if (kb < l0) { segp = s0 + kb; stride = l0; }
    else if (kb < l01) { segp = s1 + (kb - l0); stride = l1; }
    else { segp = s2 + (kb - l01); stride = l2; }
    if (tid < 128) {  // A: 32 rows x 4 vec
      int b = tid >> 2, off = (tid & 3) * 8;
      cp16(sA + tid * 8, segp + (size_t)b * stride + off);
    }
    if (!isq) {  // gates: 64 W rows x 4 vec
      int j = tid >> 2, off = (tid & 3) * 8;
      int n = (j >> 4) * 1024 + c0 + (j & 15);
      const unsigned short* src = Wc + (size_t)n * K + kb + off;
      cp16(sW + tid * 8, src);
      if (kb + 32 < K) __builtin_prefetch(src + 32, 0, 0);
    } else {  // q: 128 W rows x 4 vec
#pragma unroll
      for (int it = 0; it < 2; ++it) {
        int i = tid + it * 256;
        int n = i >> 2, off = (i & 3) * 8;
        const unsigned short* src = Wq + (size_t)n * K + kb + off;
        cp16(sW + i * 8, src);
        if (kb + 32 < K) __builtin_prefetch(src + 32, 0, 0);
      }
    }
    async_fence();
    __syncthreads();
    if (!isq) {
      int mt = wave & 1, g = wave >> 1;
      v16bf a = load_a_frag(sA + mt * 512);
      v16bf b = load_b_frag(sW + g * 512);
      acc0 = wmma_bf(a, b, acc0);
    } else {
      v16bf b = load_b_frag(sW + wave * 512);
      v16bf a0 = load_a_frag(sA);
      v16bf a1 = load_a_frag(sA + 512);
      acc0 = wmma_bf(a0, b, acc0);
      acc1 = wmma_bf(a1, b, acc1);
    }
    __syncthreads();
  }
  if (!isq) {
    int mt = wave & 1, g = wave >> 1;
    int cell = lane & 15;
    int mb = mt * 16 + ((lane >> 4) << 3);
    int n = g * 1024 + c0 + cell;
#pragma unroll
    for (int v = 0; v < 8; ++v)
      gbuf[(mb + v) * 64 + g * 16 + cell] = acc0[v] + bias[n];
    __syncthreads();
    for (int i = tid; i < 512; i += 256) {
      int b = i >> 4, cell2 = i & 15;
      int j = c0 + cell2;
      float ig = gbuf[b * 64 + cell2];
      float fg = gbuf[b * 64 + 16 + cell2];
      float gg = gbuf[b * 64 + 32 + cell2];
      float og = gbuf[b * 64 + 48 + cell2];
      float co = cs[b * 1024 + j];
      float cn = sigm(fg) * co + sigm(ig) * tanhf(gg);
      float h = sigm(og) * tanhf(cn);
      cs[b * 1024 + j] = cn;
      hs[b * 1024 + j] = h;
      hbf_out[b * 1024 + j] = f2bf_bits(h);
    }
  } else {
    int nn = wave * 16 + (lane & 15);
    int mb0 = (lane >> 4) << 3;
#pragma unroll
    for (int v = 0; v < 8; ++v) {
      qout[(mb0 + v) * 128 + nn] = acc0[v];
      qout[(16 + mb0 + v) * 128 + nn] = acc1[v];
    }
  }
}

// ------------------- location attention: conv + WMMA + energies ------------
__global__ void k_attn_energy(const float* __restrict__ cum, const float* __restrict__ cw,
                              const unsigned short* __restrict__ Wl,
                              const float* __restrict__ q, const float* __restrict__ pe,
                              const float* __restrict__ al,
                              const unsigned char* __restrict__ mask,
                              float* __restrict__ e) {
  __shared__ __align__(16) unsigned short sA[4 * 512];
  __shared__ __align__(16) unsigned short sW[128 * 32];
  int tid = threadIdx.x, wave = tid >> 5, lane = tid & 31;
#pragma unroll
  for (int it = 0; it < 2; ++it) {  // loc_lin [128][32]: 512 vecs of 16B
    int i = tid + it * 256;
    cp16(sW + i * 8, Wl + i * 8);
  }
  int r0 = blockIdx.x * 64 + wave * 16;
  for (int j = 0; j < 16; ++j) {
    int r = r0 + j, b = r / 200, s = r % 200;
    float a = 0.0f;
#pragma unroll
    for (int tp = 0; tp < 31; ++tp) {
      int ss = s + tp - 15;
      if (ss >= 0 && ss < 200) a += cum[b * 200 + ss] * cw[lane * 31 + tp];
    }
    sA[wave * 512 + j * 32 + lane] = f2bf_bits(a);
  }
  async_fence();
  __syncthreads();
  v16bf a = load_a_frag(sA + wave * 512);
  float ep[8] = {0.f, 0.f, 0.f, 0.f, 0.f, 0.f, 0.f, 0.f};
  for (int nt = 0; nt < 8; ++nt) {
    v16bf b = load_b_frag(sW + nt * 512);
    v8f acc = {0.f, 0.f, 0.f, 0.f, 0.f, 0.f, 0.f, 0.f};
    acc = wmma_bf(a, b, acc);
    int n = nt * 16 + (lane & 15);
    float aln = al[n];
#pragma unroll
    for (int v = 0; v < 8; ++v) {
      int r = r0 + ((lane >> 4) << 3) + v;
      int bb = r / 200;
      float x = acc[v] + q[bb * 128 + n] + pe[(size_t)r * 128 + n];
      ep[v] += tanhf(x) * aln;
    }
  }
#pragma unroll
  for (int v = 0; v < 8; ++v) {
    float s = ep[v];
    s += __shfl_xor(s, 1, 16);
    s += __shfl_xor(s, 2, 16);
    s += __shfl_xor(s, 4, 16);
    s += __shfl_xor(s, 8, 16);
    if ((lane & 15) == 0) {
      int r = r0 + ((lane >> 4) << 3) + v;
      e[r] = mask[r] ? s : -1e30f;
    }
  }
}

// -------- softmax over S, context, cum update, mel/stop projections --------
__global__ void k_attn_ctx(const float* __restrict__ e, float* __restrict__ cum,
                           const float* __restrict__ enc, const float* __restrict__ h2,
                           float* __restrict__ ctx, unsigned short* __restrict__ ctxbf,
                           const float* __restrict__ pw, const float* __restrict__ pb,
                           const float* __restrict__ sw, const float* __restrict__ sb,
                           float* __restrict__ omel, float* __restrict__ ostop, int t) {
  __shared__ float as_[200];
  __shared__ float red[256];
  __shared__ float cs_[512];
  int b = blockIdx.x, tid = threadIdx.x;
  float m = -1e30f;
  for (int s = tid; s < 200; s += 256) {
    float v = e[b * 200 + s];
    as_[s] = v;
    m = fmaxf(m, v);
  }
  red[tid] = m;
  __syncthreads();
  for (int o = 128; o > 0; o >>= 1) {
    if (tid < o) red[tid] = fmaxf(red[tid], red[tid + o]);
    __syncthreads();
  }
  m = red[0];
  __syncthreads();
  float sum = 0.0f;
  for (int s = tid; s < 200; s += 256) {
    float v = expf(as_[s] - m);
    as_[s] = v;
    sum += v;
  }
  red[tid] = sum;
  __syncthreads();
  for (int o = 128; o > 0; o >>= 1) {
    if (tid < o) red[tid] += red[tid + o];
    __syncthreads();
  }
  float inv = 1.0f / red[0];
  __syncthreads();
  for (int s = tid; s < 200; s += 256) {
    float a = as_[s] * inv;
    as_[s] = a;
    cum[b * 200 + s] += a;
  }
  __syncthreads();
  for (int d = tid; d < 512; d += 256) {
    float acc = 0.0f;
    for (int s = 0; s < 200; ++s) acc += as_[s] * enc[((size_t)b * 200 + s) * 512 + d];
    ctx[b * 512 + d] = acc;
    ctxbf[b * 512 + d] = f2bf_bits(acc);
    cs_[d] = acc;
  }
  __syncthreads();
  for (int j = tid; j < 81; j += 256) {
    const float* w = (j < 80) ? (pw + (size_t)j * 1536) : sw;
    float acc = (j < 80) ? pb[j] : sb[0];
    for (int k = 0; k < 1024; ++k) acc += h2[b * 1024 + k] * w[k];
    for (int k = 0; k < 512; ++k) acc += cs_[k] * w[1024 + k];
    if (j < 80) omel[((size_t)b * 80 + j) * 800 + t] = acc;
    else ostop[b * 800 + t] = acc;
  }
}

// ------------- postnet conv (K=Cin*5, tap-major repacked W) + BN -----------
// Wt is repacked [Cout][5][Cin] so K'=tap*Cin+c is contiguous per row.
__global__ void k_postconv(const unsigned short* __restrict__ X, int Cin,
                           const unsigned short* __restrict__ Wt, int Cout,
                           const float* __restrict__ scale, const float* __restrict__ shift,
                           int act_tanh, unsigned short* __restrict__ Y, int T) {
  __shared__ __align__(16) unsigned short sA[32 * 32];
  __shared__ __align__(16) unsigned short sW[64 * 32];
  int tid = threadIdx.x, wave = tid >> 5, lane = tid & 31;
  int mt = wave & 1, nt = wave >> 1;
  int m0 = blockIdx.y * 32, n0 = blockIdx.x * 64;
  int K = Cin * 5;
  v8f acc = {0.f, 0.f, 0.f, 0.f, 0.f, 0.f, 0.f, 0.f};
  int KC = (K + 31) >> 5;
  for (int kc = 0; kc < KC; ++kc) {
    int kb = kc << 5;
    // W staging (contiguous thanks to tap-major repack)
    if (kb + 32 <= K) {
      int r = tid >> 2, off = (tid & 3) * 8;
      int n = n0 + r;
      if (n < Cout) cp16(sW + tid * 8, Wt + (size_t)n * K + kb + off);
      else zero16(sW + tid * 8);
    } else {
      for (int i = tid; i < 2048; i += 256) {
        int r = i >> 5, kk = kb + (i & 31);
        int n = n0 + r;
        sW[i] = (kk < K && n < Cout) ? Wt[(size_t)n * K + kk] : 0;
      }
    }
    // A staging: chunk stays within one tap iff Cin % 32 == 0
    if ((Cin & 31) == 0) {
      if (tid < 128) {
        int r = tid >> 2, off = (tid & 3) * 8;
        int tap = kb / Cin, cc = (kb % Cin) + off;
        int row = m0 + r, bb = row / T, tt = row % T;
        int ts = tt + tap - 2;
        if (ts >= 0 && ts < T) cp16(sA + tid * 8, X + ((size_t)bb * T + ts) * Cin + cc);
        else zero16(sA + tid * 8);
      }
    } else {
      for (int i = tid; i < 1024; i += 256) {
        int r = i >> 5, kk = kb + (i & 31);
        unsigned short v = 0;
        if (kk < K) {
          int tap = kk / Cin, c = kk % Cin;
          int row = m0 + r, bb = row / T, tt = row % T;
          int ts = tt + tap - 2;
          if (ts >= 0 && ts < T) v = X[((size_t)bb * T + ts) * Cin + c];
        }
        sA[i] = v;
      }
    }
    async_fence();
    __syncthreads();
    v16bf a = load_a_frag(sA + mt * 512);
    v16bf b = load_b_frag(sW + nt * 512);
    acc = wmma_bf(a, b, acc);
    __syncthreads();
  }
  int gn = n0 + nt * 16 + (lane & 15);
  int mb = m0 + mt * 16 + ((lane >> 4) << 3);
  if (gn < Cout) {
#pragma unroll
    for (int v = 0; v < 8; ++v) {
      float x = acc[v] * scale[gn] + shift[gn];
      if (act_tanh) x = tanhf(x);
      Y[(size_t)(mb + v) * Cout + gn] = f2bf_bits(x);
    }
  }
}

// ---------------------------------------------------------------------------

extern "C" void kernel_launch(void* const* d_in, const int* in_sizes, int n_in,
                              void* d_out, int out_size, void* d_ws, size_t ws_size,
                              hipStream_t stream) {
  (void)in_sizes; (void)n_in; (void)out_size; (void)ws_size;
  const int B = 32, T = 800, S = 200;

  const float* mels = (const float*)d_in[0];
  const float* enc  = (const float*)d_in[1];
  const unsigned char* mask = (const unsigned char*)d_in[2];
  int ip = 3;
  const float* pre_w1 = (const float*)d_in[ip++];
  const float* pre_b1 = (const float*)d_in[ip++];
  const float* pre_w2 = (const float*)d_in[ip++];
  const float* pre_b2 = (const float*)d_in[ip++];
  const float* enc_w  = (const float*)d_in[ip++];
  const float* enc_b  = (const float*)d_in[ip++];
  const float* Wih0   = (const float*)d_in[ip++];
  const float* Whh0   = (const float*)d_in[ip++];
  const float* b0     = (const float*)d_in[ip++];
  const float* Wih1   = (const float*)d_in[ip++];
  const float* Whh1   = (const float*)d_in[ip++];
  const float* b1     = (const float*)d_in[ip++];
  const float* loc_conv = (const float*)d_in[ip++];
  const float* loc_lin  = (const float*)d_in[ip++];
  const float* rnn_lin  = (const float*)d_in[ip++];
  const float* align_lin = (const float*)d_in[ip++];
  const float* proj_w = (const float*)d_in[ip++];
  const float* proj_b = (const float*)d_in[ip++];
  const float* stop_w = (const float*)d_in[ip++];
  const float* stop_b = (const float*)d_in[ip++];
  const float* post_lin_w = (const float*)d_in[ip++];
  const float* post_lin_b = (const float*)d_in[ip++];
  const float* pconv_f[5]; const float* bng[5]; const float* bnb[5];
  const float* bnm[5]; const float* bnv[5];
  for (int i = 0; i < 5; ++i) {
    pconv_f[i] = (const float*)d_in[ip++];
    bng[i] = (const float*)d_in[ip++];
    bnb[i] = (const float*)d_in[ip++];
    bnm[i] = (const float*)d_in[ip++];
    bnv[i] = (const float*)d_in[ip++];
  }

  float* out_mel0 = (float*)d_out;
  float* out_mel1 = out_mel0 + (size_t)B * 80 * T;
  float* out_stop = out_mel1 + (size_t)B * 80 * T;

  char* base = (char*)d_ws;
  size_t off = 0;
  auto A16 = [&](size_t n) -> unsigned short* {
    unsigned short* p = (unsigned short*)(base + off);
    off = (off + n * 2 + 255) & ~(size_t)255;
    return p;
  };
  auto A32 = [&](size_t n) -> float* {
    float* p = (float*)(base + off);
    off = (off + n * 4 + 255) & ~(size_t)255;
    return p;
  };
  unsigned short* W0cat = A16((size_t)4096 * 1792);
  unsigned short* W1cat = A16((size_t)4096 * 2048);
  unsigned short* prew1b = A16(256 * 80);
  unsigned short* prew2b = A16(256 * 256);
  unsigned short* encwb  = A16(128 * 512);
  unsigned short* loclinb = A16(128 * 32);
  unsigned short* rnnlinb = A16(128 * 2048);
  unsigned short* plinwb  = A16(80 * 512);
  unsigned short* pconvb[5];
  float* bnscale[5]; float* bnshift[5];
  pconvb[0] = A16((size_t)512 * 80 * 5);
  for (int i = 1; i < 5; ++i) pconvb[i] = A16((size_t)512 * 512 * 5);
  for (int i = 0; i < 5; ++i) { bnscale[i] = A32(512); bnshift[i] = A32(512); }
  unsigned short* X0 = A16((size_t)B * T * 80);
  unsigned short* P1 = A16((size_t)B * T * 256);
  unsigned short* P  = A16((size_t)B * T * 256);
  unsigned short* encbf = A16((size_t)B * S * 512);
  float* proc = A32((size_t)B * S * 128);
  float* h1 = A32(B * 1024); float* c1 = A32(B * 1024);
  float* h2 = A32(B * 1024); float* c2 = A32(B * 1024);
  float* ctx = A32(B * 512); float* cum = A32(B * S);
  float* qv = A32(B * 128);  float* ev = A32(B * S);
  unsigned short* h1b[2] = {A16(B * 1024), A16(B * 1024)};
  unsigned short* h2b[2] = {A16(B * 1024), A16(B * 1024)};
  unsigned short* ctxb = A16(B * 512);
  unsigned short* melb = A16((size_t)B * T * 80);
  unsigned short* pa = A16((size_t)B * T * 512);
  unsigned short* pb = A16((size_t)B * T * 512);
  float* ptmp = A32((size_t)B * T * 80);

  auto g1 = [](int n) { return dim3((n + 255) / 256); };

  // ---- weight prep ----
  k_f2b<<<g1(256 * 80), 256, 0, stream>>>(pre_w1, prew1b, 256 * 80);
  k_f2b<<<g1(256 * 256), 256, 0, stream>>>(pre_w2, prew2b, 256 * 256);
  k_f2b<<<g1(128 * 512), 256, 0, stream>>>(enc_w, encwb, 128 * 512);
  k_f2b<<<g1(128 * 32), 256, 0, stream>>>(loc_lin, loclinb, 128 * 32);
  k_f2b<<<g1(128 * 2048), 256, 0, stream>>>(rnn_lin, rnnlinb, 128 * 2048);
  k_f2b<<<g1(80 * 512), 256, 0, stream>>>(post_lin_w, plinwb, 80 * 512);
  k_convrepack<<<g1(512 * 80 * 5), 256, 0, stream>>>(pconv_f[0], pconvb[0], 512, 80);
  for (int i = 1; i < 5; ++i)
    k_convrepack<<<g1(512 * 512 * 5), 256, 0, stream>>>(pconv_f[i], pconvb[i], 512, 512);
  k_concat2<<<g1(4096 * 1792), 256, 0, stream>>>(Wih0, 768, Whh0, 1024, W0cat, 4096);
  k_concat2<<<g1(4096 * 2048), 256, 0, stream>>>(Wih1, 1024, Whh1, 1024, W1cat, 4096);
  for (int i = 0; i < 5; ++i)
    k_bnfold<<<g1(512), 256, 0, stream>>>(bng[i], bnb[i], bnm[i], bnv[i],
                                          bnscale[i], bnshift[i], 512);

  // ---- zero recurrent state ----
  k_zero_f<<<g1(B * 1024), 256, 0, stream>>>(h1, B * 1024);
  k_zero_f<<<g1(B * 1024), 256, 0, stream>>>(c1, B * 1024);
  k_zero_f<<<g1(B * 1024), 256, 0, stream>>>(h2, B * 1024);
  k_zero_f<<<g1(B * 1024), 256, 0, stream>>>(c2, B * 1024);
  k_zero_f<<<g1(B * 512), 256, 0, stream>>>(ctx, B * 512);
  k_zero_f<<<g1(B * S), 256, 0, stream>>>(cum, B * S);
  for (int i = 0; i < 2; ++i) {
    k_zero_h<<<g1(B * 1024), 256, 0, stream>>>(h1b[i], B * 1024);
    k_zero_h<<<g1(B * 1024), 256, 0, stream>>>(h2b[i], B * 1024);
  }
  k_zero_h<<<g1(B * 512), 256, 0, stream>>>(ctxb, B * 512);

  // ---- prenet ----
  k_mels_to_rows<<<g1(B * 80 * T), 256, 0, stream>>>(mels, X0, B, 80, T);
  k_gemm<<<dim3(4, 800), 256, 0, stream>>>(X0, prew1b, pre_b1, nullptr, P1,
                                           B * T, 256, 80, 1);
  k_gemm<<<dim3(4, 800), 256, 0, stream>>>(P1, prew2b, pre_b2, nullptr, P,
                                           B * T, 256, 256, 1);
  // ---- encoder projection ----
  k_f2b<<<g1(B * S * 512), 256, 0, stream>>>(enc, encbf, B * S * 512);
  k_gemm<<<dim3(2, 200), 256, 0, stream>>>(encbf, encwb, enc_b, proc, nullptr,
                                           B * S, 128, 512, 0);

  // ---- 800-step sequential decode ----
  for (int t = 0; t < T; ++t) {
    int p = t & 1;
    k_lstm_step<<<64, 256, 0, stream>>>(P + (size_t)t * B * 256, 256,
                                        ctxb, 512, h1b[p], 1024,
                                        W0cat, 1792, b0, c1, h1, h1b[1 - p],
                                        nullptr, nullptr);
    k_lstm_step<<<65, 256, 0, stream>>>(h1b[1 - p], 1024, h2b[p], 1024,
                                        nullptr, 0,
                                        W1cat, 2048, b1, c2, h2, h2b[1 - p],
                                        rnnlinb, qv);
    k_attn_energy<<<100, 256, 0, stream>>>(cum, loc_conv, loclinb, qv, proc,
                                           align_lin, mask, ev);
    k_attn_ctx<<<32, 256, 0, stream>>>(ev, cum, enc, h2, ctx, ctxb,
                                       proj_w, proj_b, stop_w, stop_b,
                                       out_mel0, out_stop, t);
  }

  // ---- postnet ----
  k_mel_to_post<<<g1(B * 80 * T), 256, 0, stream>>>(out_mel0, melb, B, 80, T);
  k_postconv<<<dim3(8, 800), 256, 0, stream>>>(melb, 80, pconvb[0], 512,
                                               bnscale[0], bnshift[0], 1, pa, T);
  k_postconv<<<dim3(8, 800), 256, 0, stream>>>(pa, 512, pconvb[1], 512,
                                               bnscale[1], bnshift[1], 1, pb, T);
  k_postconv<<<dim3(8, 800), 256, 0, stream>>>(pb, 512, pconvb[2], 512,
                                               bnscale[2], bnshift[2], 1, pa, T);
  k_postconv<<<dim3(8, 800), 256, 0, stream>>>(pa, 512, pconvb[3], 512,
                                               bnscale[3], bnshift[3], 1, pb, T);
  k_postconv<<<dim3(8, 800), 256, 0, stream>>>(pb, 512, pconvb[4], 512,
                                               bnscale[4], bnshift[4], 0, pa, T);
  k_gemm<<<dim3(2, 800), 256, 0, stream>>>(pa, plinwb, post_lin_b, ptmp, nullptr,
                                           B * T, 80, 512, 0);
  k_residual<<<g1(B * 80 * T), 256, 0, stream>>>(out_mel0, ptmp, out_mel1, B, 80, T);
}
